// ShortLSTM_GAT_46428596470350
// MI455X (gfx1250) — matrix operand order repaired and checked by reading, not compiled, not run on hardware
//
#include <hip/hip_runtime.h>

// ============================================================================
// ShortLSTM + GAT pipeline for MI455X (gfx1250, wave32, WMMA, TDM).
//
// Kernel 1 (compute-bound): 2-layer LSTM over T=24 steps, 16 nodes per wave.
//   - h@W matmuls via v_wmma_f32_16x16x32_f16 (96 WMMA / wave / step).
//   - A fragments loaded as 2x ds_load_b128 (16-bit A layout = two contiguous
//     16B runs per lane); B fragments pre-swizzled once into LDS.
//   - Branchless activations: v_exp/v_rcp sigmoid + native v_tanh_f32.
//   - Input tile staged via TENSOR_LOAD_TO_LDS (TDM) + s_wait_tensorcnt, then
//     transposed once so per-step x reads are 2x ds_load_b128, no addr math.
// Kernels 2..7 (memory/atomic-bound): segment softmax + scatter-add GAT,
//   ELU + final linear.
// ============================================================================

typedef __attribute__((ext_vector_type(16))) _Float16 v16h;
typedef __attribute__((ext_vector_type(8)))  _Float16 v8h;
typedef __attribute__((ext_vector_type(8)))  float    v8f;
typedef unsigned __attribute__((ext_vector_type(4)))  u32x4;
typedef int      __attribute__((ext_vector_type(4)))  i32x4;
typedef int      __attribute__((ext_vector_type(8)))  i32x8;

static constexpr int   kN     = 100000;
static constexpr int   kT     = 24;
static constexpr int   kHeads = 4;
static constexpr int   kE     = 1600000;
static constexpr float kSlope = 0.2f;

// ---------------- LDS layout for the LSTM kernel (bytes) --------------------
static constexpr int FRAG_HALFS = 512;                       // 32 lanes * 16 f16
static constexpr int OFF_WHH1 = 0;                           // 32 frags (2 kc * 16 nt)
static constexpr int OFF_WIH2 = OFF_WHH1 + 32 * FRAG_HALFS * 2;
static constexpr int OFF_WHH2 = OFF_WIH2 + 32 * FRAG_HALFS * 2;
static constexpr int OFF_GATW = OFF_WHH2 + 32 * FRAG_HALFS * 2;  // 8 frags
static constexpr int OFF_B1   = OFF_GATW + 8 * FRAG_HALFS * 2;
static constexpr int OFF_B2   = OFF_B1 + 256 * 4;
static constexpr int OFF_WI1  = OFF_B2 + 256 * 4;
static constexpr int OFF_WAVE = OFF_WI1 + 256 * 4;
// per-wave: h1 dbl(4096) + h2 dbl(4096) + x row-major(1536) + x transposed(1536)
static constexpr int PER_WAVE = 4096 + 4096 + 1536 + 1536;
static constexpr int LDS_TOTAL = OFF_WAVE + 8 * PER_WAVE;    // ~195 KB < 320 KB WGP LDS

// ---------------------------- device helpers --------------------------------
__device__ __forceinline__ v8f wmma16(v16h a, v16h b, v8f c) {
  // D = A(16x32 f16) x B(32x16 f16) + C(16x16 f32)
  return __builtin_amdgcn_wmma_f32_16x16x32_f16(false, a, false, b,
                                                (short)0, c, false, false);
}

// B fragment: stored pre-swizzled, lane-contiguous (32B per lane).
__device__ __forceinline__ v16h ldB(const _Float16* frags, int frag, int lane) {
  return *(const v16h*)(frags + frag * FRAG_HALFS + lane * 16);
}

// A fragment from row-major 16x64 f16 LDS buffer.
// ISA A-layout (16-bit, 16x32): lane l<16 holds row l, K=0..7 (V0-3) and
// K=16..23 (V4-7); lane l>=16 holds row l-16, K=8..15 / K=24..31.  Each half
// is a CONTIGUOUS 16-byte run -> two ds_load_b128 per fragment.
__device__ __forceinline__ v16h ldA(const _Float16* h, int lane, int kOff) {
  const int row = lane & 15;
  const int hi  = (lane >> 4) & 1;
  const _Float16* p = h + row * 64 + kOff + hi * 8;   // 16B aligned
  union { v16h v; v8h q[2]; } r;
  r.q[0] = *(const v8h*)(p);
  r.q[1] = *(const v8h*)(p + 16);
  return r.v;
}

// Swizzle row-major W (J x 64), B[k][j] = W[j*64+k], into B-fragment order.
__device__ __forceinline__ void fillB(_Float16* frags, const float* W, int J, int tid) {
  const int total = J * 64;
  const int nJT = J >> 4;
  for (int idx = tid; idx < total; idx += 256) {
    const int j = idx >> 6, k = idx & 63;
    const int kc = k >> 5, kl = k & 31;
    const int jt = j >> 4, jl = j & 15;
    const int lane = jl + ((kl & 8) ? 16 : 0);
    const int d = ((kl & 7) | ((kl & 16) >> 1)) >> 1;
    const int half = kl & 1;
    frags[(kc * nJT + jt) * FRAG_HALFS + lane * 16 + d * 2 + half] = (_Float16)W[idx];
  }
}

// Branchless sigmoid: v_exp_f32 + v_rcp_f32 (no divergent range reduction).
__device__ __forceinline__ float sigf(float x) {
#if __has_builtin(__builtin_amdgcn_rcpf)
  return __builtin_amdgcn_rcpf(1.0f + __expf(-x));
#else
  return 1.0f / (1.0f + __expf(-x));
#endif
}
// Native CDNA5 v_tanh_f32 when available, else branchless 2*sig(2x)-1.
__device__ __forceinline__ float tanh_fast(float x) {
#if __has_builtin(__builtin_amdgcn_tanhf)
  return __builtin_amdgcn_tanhf(x);
#else
  return 2.0f * sigf(2.0f * x) - 1.0f;
#endif
}

// Monotonic int key for float atomic-max.
__device__ __forceinline__ unsigned fflip(float f) {
  unsigned u = __float_as_uint(f);
  unsigned mask = (unsigned)(-(int)(u >> 31)) | 0x80000000u;
  return u ^ mask;
}
__device__ __forceinline__ float funflip(unsigned u) {
  unsigned mask = (u >> 31) ? 0x80000000u : 0xFFFFFFFFu;
  return __uint_as_float(u ^ mask);
}

// ============================================================================
// Kernel 1: fused 2-layer LSTM (T steps) + GAT input projection, 16 nodes/wave
// ============================================================================
__global__ void __launch_bounds__(256)
k_lstm_gat_nodes(const float* __restrict__ xin,
                 const float* __restrict__ Wih1, const float* __restrict__ Whh1,
                 const float* __restrict__ bih1, const float* __restrict__ bhh1,
                 const float* __restrict__ Wih2, const float* __restrict__ Whh2,
                 const float* __restrict__ bih2, const float* __restrict__ bhh2,
                 const float* __restrict__ gatW, const float* __restrict__ attSrc,
                 const float* __restrict__ attDst,
                 float* __restrict__ xfeat, float* __restrict__ asrc,
                 float* __restrict__ adst) {
  extern __shared__ char smem[];
  const int tid = threadIdx.x;
  const int wave = tid >> 5;
  const int lane = tid & 31;

  _Float16* fWhh1 = (_Float16*)(smem + OFF_WHH1);
  _Float16* fWih2 = (_Float16*)(smem + OFF_WIH2);
  _Float16* fWhh2 = (_Float16*)(smem + OFF_WHH2);
  _Float16* fGat  = (_Float16*)(smem + OFF_GATW);
  float* b1   = (float*)(smem + OFF_B1);
  float* b2   = (float*)(smem + OFF_B2);
  float* wi1  = (float*)(smem + OFF_WI1);

  fillB(fWhh1, Whh1, 256, tid);
  fillB(fWih2, Wih2, 256, tid);
  fillB(fWhh2, Whh2, 256, tid);
  fillB(fGat,  gatW,  64, tid);
  if (tid < 256) {
    b1[tid]  = bih1[tid] + bhh1[tid];
    b2[tid]  = bih2[tid] + bhh2[tid];
    wi1[tid] = Wih1[tid];
  }

  char* wb = smem + OFF_WAVE + wave * PER_WAVE;
  _Float16* h1buf = (_Float16*)wb;            // 2 x (16x64) f16
  _Float16* h2buf = (_Float16*)(wb + 4096);   // 2 x (16x64) f16
  float*    xst   = (float*)(wb + 8192);      // 16 x 24 f32 (row-major, TDM dst)
  float*    xstT  = (float*)(wb + 8192 + 1536); // 24 x 16 f32 (transposed)

  const int nb = blockIdx.x * 128 + wave * 16;  // first node of this wave

  for (int i = lane; i < 2048; i += 32) { h1buf[i] = (_Float16)0.f; h2buf[i] = (_Float16)0.f; }

  // ---- stage the wave's 16x24 input tile: TDM tensor load (global -> LDS) ----
#if __has_builtin(__builtin_amdgcn_tensor_load_to_lds)
  {
    const unsigned long long ga =
        (unsigned long long)(uintptr_t)(xin + (size_t)nb * kT);
    const unsigned ldsOff = (unsigned)((char*)xst - (char*)smem);
    // D# group 0: count=1 | lds_addr | global_addr[56:0] | type=2
    u32x4 g0;
    g0.x = 1u;
    g0.y = ldsOff;
    g0.z = (unsigned)(ga & 0xffffffffu);
    g0.w = (unsigned)((ga >> 32) & 0x1ffffffu) | (2u << 30);
    // D# group 1: data_size=4B; tensor_dim0=24; tensor_dim1=N; tile 24x16;
    // tensor_dim0_stride=24 (row-major (N,T) input)
    i32x8 g1;
    g1[0] = 0x00020000;                                   // data_size=2 (4B)
    g1[1] = (kT << 16);                                   // tensor_dim0 lo16
    g1[2] = (int)((kN & 0xffff) << 16);                   // dim0 hi16=0 | dim1 lo16
    g1[3] = (kT << 16) | (kN >> 16);                      // dim1 hi16 | tile_dim0
    g1[4] = 16;                                           // tile_dim1=16, tile_dim2=0
    g1[5] = kT;                                           // dim0_stride lo32
    g1[6] = 0;
    g1[7] = 0;
    const i32x4 gz = {0, 0, 0, 0};
#if defined(__clang_major__) && (__clang_major__ >= 23)
    const i32x8 gz8 = {0, 0, 0, 0, 0, 0, 0, 0};
    __builtin_amdgcn_tensor_load_to_lds(g0, g1, gz, gz, gz8, 0);
#else
    __builtin_amdgcn_tensor_load_to_lds(g0, g1, gz, gz, 0);
#endif
  }
  __syncthreads();
#if __has_builtin(__builtin_amdgcn_s_wait_tensorcnt)
  __builtin_amdgcn_s_wait_tensorcnt(0);
#endif
  // one-time transpose: (node,t) -> (t,node), per-step reads become b128s
  for (int i = lane; i < 16 * kT; i += 32) {
    const int r = i & 15, t = i >> 4;
    xstT[i] = xst[r * kT + t];
  }
#else
  for (int i = lane; i < 16 * kT; i += 32) {
    const int r = i & 15, t = i >> 4;
    int n = nb + r; if (n >= kN) n = kN - 1;
    xstT[i] = xin[n * kT + t];
  }
  __syncthreads();
#endif

  const int hi  = lane >> 4;
  const int col = lane & 15;
  const v8f z8 = {0.f, 0.f, 0.f, 0.f, 0.f, 0.f, 0.f, 0.f};
  v8f c1s[4], c2s[4];
#pragma unroll
  for (int i = 0; i < 4; ++i) { c1s[i] = z8; c2s[i] = z8; }

  const float4* xtp = (const float4*)xstT;  // [kT*4] float4s
  int cur = 0;
  for (int t = 0; t < kT; ++t) {
    union { float4 f4[2]; float f[8]; } xr;
    xr.f4[0] = xtp[t * 4 + hi * 2];          // nodes hi*8 .. hi*8+3
    xr.f4[1] = xtp[t * 4 + hi * 2 + 1];      // nodes hi*8+4 .. hi*8+7

    // ---- layer 1: z1 = h1 @ Whh1^T + x*Wih1 + b1 ----
    {
      const v16h a0 = ldA(h1buf + cur * 1024, lane, 0);
      const v16h a1 = ldA(h1buf + cur * 1024, lane, 32);
#pragma unroll
      for (int nt = 0; nt < 4; ++nt) {
        v8f zg[4];
#pragma unroll
        for (int g = 0; g < 4; ++g) {
          const int ct = g * 4 + nt;                  // z column tile
          v8f acc = z8;
          acc = wmma16(a0, ldB(fWhh1, ct, lane), acc);
          acc = wmma16(a1, ldB(fWhh1, 16 + ct, lane), acc);
          const int cg = g * 64 + nt * 16 + col;
          const float bb = b1[cg];
          const float wi = wi1[cg];
#pragma unroll
          for (int v = 0; v < 8; ++v) acc[v] += bb + xr.f[v] * wi;
          zg[g] = acc;
        }
#pragma unroll
        for (int v = 0; v < 8; ++v) {
          const float i_ = sigf(zg[0][v]);
          const float f_ = sigf(zg[1][v]);
          const float g_ = tanh_fast(zg[2][v]);
          const float o_ = sigf(zg[3][v]);
          const float c  = f_ * c1s[nt][v] + i_ * g_;
          c1s[nt][v] = c;
          const int row = v + 8 * hi;
          h1buf[(cur ^ 1) * 1024 + row * 64 + nt * 16 + col] =
              (_Float16)(o_ * tanh_fast(c));
        }
      }
    }
    // ---- layer 2: z2 = h1_new @ Wih2^T + h2 @ Whh2^T + b2 ----
    {
      const v16h ax0 = ldA(h1buf + (cur ^ 1) * 1024, lane, 0);
      const v16h ax1 = ldA(h1buf + (cur ^ 1) * 1024, lane, 32);
      const v16h ah0 = ldA(h2buf + cur * 1024, lane, 0);
      const v16h ah1 = ldA(h2buf + cur * 1024, lane, 32);
#pragma unroll
      for (int nt = 0; nt < 4; ++nt) {
        v8f zg[4];
#pragma unroll
        for (int g = 0; g < 4; ++g) {
          const int ct = g * 4 + nt;
          v8f acc = z8;
          acc = wmma16(ax0, ldB(fWih2, ct, lane), acc);
          acc = wmma16(ax1, ldB(fWih2, 16 + ct, lane), acc);
          acc = wmma16(ah0, ldB(fWhh2, ct, lane), acc);
          acc = wmma16(ah1, ldB(fWhh2, 16 + ct, lane), acc);
          const float bb = b2[g * 64 + nt * 16 + col];
#pragma unroll
          for (int v = 0; v < 8; ++v) acc[v] += bb;
          zg[g] = acc;
        }
#pragma unroll
        for (int v = 0; v < 8; ++v) {
          const float i_ = sigf(zg[0][v]);
          const float f_ = sigf(zg[1][v]);
          const float g_ = tanh_fast(zg[2][v]);
          const float o_ = sigf(zg[3][v]);
          const float c  = f_ * c2s[nt][v] + i_ * g_;
          c2s[nt][v] = c;
          const int row = v + 8 * hi;
          h2buf[(cur ^ 1) * 1024 + row * 64 + nt * 16 + col] =
              (_Float16)(o_ * tanh_fast(c));
        }
      }
    }
    cur ^= 1;
  }

  // ---- fused GAT input projection: x = h2 @ gatW^T (64x64), per-head dots ----
  {
    const v16h a0 = ldA(h2buf + cur * 1024, lane, 0);
    const v16h a1 = ldA(h2buf + cur * 1024, lane, 32);
#pragma unroll
    for (int nt = 0; nt < 4; ++nt) {                 // nt == head (C == 16)
      v8f acc = z8;
      acc = wmma16(a0, ldB(fGat, nt, lane), acc);
      acc = wmma16(a1, ldB(fGat, 4 + nt, lane), acc);
      const float aS = attSrc[nt * 16 + col];
      const float aD = attDst[nt * 16 + col];
#pragma unroll
      for (int v = 0; v < 8; ++v) {
        const int row = v + 8 * hi;
        const int n = nb + row;
        const float xv = acc[v];
        if (n < kN) xfeat[n * 64 + nt * 16 + col] = xv;
        float s = xv * aS, d = xv * aD;
#pragma unroll
        for (int m = 1; m < 16; m <<= 1) {
          s += __shfl_xor(s, m, 32);
          d += __shfl_xor(d, m, 32);
        }
        if (col == 0 && n < kN) { asrc[n * 4 + nt] = s; adst[n * 4 + nt] = d; }
      }
    }
  }
}

// ============================================================================
// Kernel 2: workspace init (every call; ws is poisoned by harness)
// ============================================================================
__global__ void k_init(float* __restrict__ agg, unsigned* __restrict__ amaxK,
                       float* __restrict__ denom) {
  const int i = blockIdx.x * 256 + threadIdx.x;
  if (i < kN * 64) agg[i] = 0.f;
  if (i < kN * 4) { amaxK[i] = fflip(-1e30f); denom[i] = 0.f; }
}

// Kernel 3: per-head edge scalar s_h = sum_c gat_lin_edge_w[h,c]*att_edge[h,c]
__global__ void k_sedge(const float* __restrict__ gelw, const float* __restrict__ atte,
                        float* __restrict__ sedge) {
  const int h = threadIdx.x;
  if (h < kHeads) {
    float s = 0.f;
    for (int c = 0; c < 16; ++c) s += gelw[h * 16 + c] * atte[h * 16 + c];
    sedge[h] = s;
  }
}

__device__ __forceinline__ float edge_alpha(const float* asrc, const float* adst,
                                            int s, int d, int h, float ea,
                                            const float* sedge) {
  const float a = asrc[s * 4 + h] + adst[d * 4 + h] + ea * sedge[h];
  return a > 0.f ? a : kSlope * a;
}

// Kernel 4: segment max of alpha over dst (int-key float atomic max)
__global__ void k_amax(const int* __restrict__ ei, const float* __restrict__ eattr,
                       const int* __restrict__ knn, const float* __restrict__ asrc,
                       const float* __restrict__ adst, const float* __restrict__ sedge,
                       unsigned* __restrict__ amaxK) {
  const int e = blockIdx.x * 256 + threadIdx.x;
  if (e >= kE || !knn[e]) return;
  const int s = ei[e], d = ei[kE + e];
  const float ea = eattr[e];
#pragma unroll
  for (int h = 0; h < kHeads; ++h)
    atomicMax(&amaxK[d * 4 + h], fflip(edge_alpha(asrc, adst, s, d, h, ea, sedge)));
}

// Kernel 5: segment sum of exp(alpha - amax)
__global__ void k_denom(const int* __restrict__ ei, const float* __restrict__ eattr,
                        const int* __restrict__ knn, const float* __restrict__ asrc,
                        const float* __restrict__ adst, const float* __restrict__ sedge,
                        const unsigned* __restrict__ amaxK, float* __restrict__ denom) {
  const int e = blockIdx.x * 256 + threadIdx.x;
  if (e >= kE || !knn[e]) return;
  const int s = ei[e], d = ei[kE + e];
  const float ea = eattr[e];
#pragma unroll
  for (int h = 0; h < kHeads; ++h) {
    const float al = edge_alpha(asrc, adst, s, d, h, ea, sedge);
    const float ex = __expf(al - funflip(amaxK[d * 4 + h]));
    atomicAdd(&denom[d * 4 + h], ex);
  }
}

// Kernel 6: weighted message scatter-add (64 lanes per edge)
__global__ void k_scatter(const int* __restrict__ ei, const float* __restrict__ eattr,
                          const int* __restrict__ knn, const float* __restrict__ asrc,
                          const float* __restrict__ adst, const float* __restrict__ sedge,
                          const unsigned* __restrict__ amaxK, const float* __restrict__ denom,
                          const float* __restrict__ xfeat, float* __restrict__ agg) {
  const int gt = blockIdx.x * 256 + threadIdx.x;
  const int e = gt >> 6;            // edge
  const int c = gt & 63;            // feature (head = c/16)
  if (e >= kE || !knn[e]) return;
  const int s = ei[e], d = ei[kE + e];
  __builtin_prefetch(&xfeat[s * 64 + c], 0, 1);   // global_prefetch on gather row
  const int h = c >> 4;
  const float al = edge_alpha(asrc, adst, s, d, h, eattr[e], sedge);
  const float ex = __expf(al - funflip(amaxK[d * 4 + h]));
  const float w = ex / fmaxf(denom[d * 4 + h], 1e-16f);
  atomicAdd(&agg[d * 64 + c], xfeat[s * 64 + c] * w);
}

// Kernel 7: out = (elu(agg + gat_bias)) @ fc_w^T + fc_b
__global__ void k_final(const float* __restrict__ agg, const float* __restrict__ gbias,
                        const float* __restrict__ fcw, const float* __restrict__ fcb,
                        float* __restrict__ out) {
  const int n = blockIdx.x * 256 + threadIdx.x;
  if (n >= kN) return;
  float acc = fcb[0];
  const float4* a4 = (const float4*)(agg + n * 64);
#pragma unroll 4
  for (int j4 = 0; j4 < 16; ++j4) {
    const float4 a = a4[j4];
#pragma unroll
    for (int q = 0; q < 4; ++q) {
      const int j = j4 * 4 + q;
      float v = (q == 0 ? a.x : q == 1 ? a.y : q == 2 ? a.z : a.w) + gbias[j];
      v = v > 0.f ? v : (__expf(v) - 1.f);
      acc += fcw[j] * v;
    }
  }
  out[n] = acc;
}

// ============================================================================
extern "C" void kernel_launch(void* const* d_in, const int* in_sizes, int n_in,
                              void* d_out, int out_size, void* d_ws, size_t ws_size,
                              hipStream_t stream) {
  const float* xin   = (const float*)d_in[0];
  const int*   ei    = (const int*)d_in[1];
  const float* eattr = (const float*)d_in[2];
  const int*   knn   = (const int*)d_in[3];
  const float* Wih1  = (const float*)d_in[4];
  const float* Whh1  = (const float*)d_in[5];
  const float* bih1  = (const float*)d_in[6];
  const float* bhh1  = (const float*)d_in[7];
  const float* Wih2  = (const float*)d_in[8];
  const float* Whh2  = (const float*)d_in[9];
  const float* bih2  = (const float*)d_in[10];
  const float* bhh2  = (const float*)d_in[11];
  const float* gatW  = (const float*)d_in[12];
  const float* gelw  = (const float*)d_in[13];
  const float* attS  = (const float*)d_in[14];
  const float* attD  = (const float*)d_in[15];
  const float* attE  = (const float*)d_in[16];
  const float* gbias = (const float*)d_in[17];
  const float* fcw   = (const float*)d_in[18];
  const float* fcb   = (const float*)d_in[19];
  float* out = (float*)d_out;

  // workspace layout (floats)
  float* wsf   = (float*)d_ws;
  float*    xfeat = wsf;                         // N*64
  float*    asrc  = xfeat + (size_t)kN * 64;     // N*4
  float*    adst  = asrc + (size_t)kN * 4;       // N*4
  unsigned* amaxK = (unsigned*)(adst + (size_t)kN * 4);  // N*4
  float*    denom = (float*)(amaxK + (size_t)kN * 4);    // N*4
  float*    agg   = denom + (size_t)kN * 4;      // N*64
  float*    sedge = agg + (size_t)kN * 64;       // 4

  k_init<<<(kN * 64 + 255) / 256, 256, 0, stream>>>(agg, amaxK, denom);
  k_sedge<<<1, 32, 0, stream>>>(gelw, attE, sedge);

  const int nBlocks = (kN + 127) / 128;   // 128 nodes / block (8 waves x 16)
  k_lstm_gat_nodes<<<nBlocks, 256, LDS_TOTAL, stream>>>(
      xin, Wih1, Whh1, bih1, bhh1, Wih2, Whh2, bih2, bhh2,
      gatW, attS, attD, xfeat, asrc, adst);

  k_amax<<<(kE + 255) / 256, 256, 0, stream>>>(ei, eattr, knn, asrc, adst, sedge, amaxK);
  k_denom<<<(kE + 255) / 256, 256, 0, stream>>>(ei, eattr, knn, asrc, adst, sedge, amaxK, denom);
  k_scatter<<<(kE * 64 + 255) / 256, 256, 0, stream>>>(ei, eattr, knn, asrc, adst, sedge,
                                                       amaxK, denom, xfeat, agg);
  k_final<<<(kN + 255) / 256, 256, 0, stream>>>(agg, gbias, fcw, fcb, out);

  (void)in_sizes; (void)n_in; (void)out_size; (void)ws_size;
}
